// PVConv_23313082483562
// MI455X (gfx1250) — matrix operand order repaired and checked by reading, not compiled, not run on hardware
//
#include <hip/hip_runtime.h>
#include <hip/hip_bf16.h>

// ---------------------------------------------------------------------------
// PVConv on MI455X (gfx1250): voxelize -> [conv3x3x3 + GN + swish] x2 ->
// trilinear devox  (+)  pointwise GEMM + GN + swish, via bf16 WMMA
// (v_wmma_f32_16x16x32_bf16) with LDS-staged weights in the conv hot loop.
// ---------------------------------------------------------------------------

#define BB     16
#define CIN    64
#define COUT   64
#define NPTS   32768
#define RR     32
#define R3     32768            // 32^3
#define GROUPS 8
#define GN_EPS 1e-5f
#define GN_CNT 262144.0f        // 8 channels * 32768 elements per (b,g)

typedef __attribute__((ext_vector_type(16))) __bf16   v16bf;
typedef __attribute__((ext_vector_type(8)))  float    v8f;
typedef __attribute__((ext_vector_type(4)))  unsigned u32x4;
typedef __attribute__((ext_vector_type(4)))  float    f32x4;

union AF { v16bf v; u32x4 q[2]; };

// ---------------------------------------------------------------------------
// 0. Transpose features [B,C,N] f32 -> featT [B,N,C] bf16 (LDS-tiled)
// ---------------------------------------------------------------------------
__global__ __launch_bounds__(256) void transpose_kernel(const float* __restrict__ f,
                                                        __bf16* __restrict__ ft)
{
    __shared__ float tile[32][33];
    const int tx = threadIdx.x & 31;
    const int ty = threadIdx.x >> 5;       // 0..7
    const int nb = blockIdx.x;             // N/32 tiles
    const int cb = blockIdx.y & 1;         // channel tile (64 = 2*32)
    const int b  = blockIdx.y >> 1;
#pragma unroll
    for (int k = 0; k < 4; ++k) {
        const int c = cb * 32 + ty + k * 8;
        tile[ty + k * 8][tx] = f[((size_t)(b * 64 + c) << 15) + (size_t)nb * 32 + tx];
    }
    __syncthreads();
#pragma unroll
    for (int k = 0; k < 4; ++k) {
        const int n = nb * 32 + ty + k * 8;
        ft[(((size_t)b << 15) + n) * 64 + cb * 32 + tx] = (__bf16)tile[tx][ty + k * 8];
    }
}

// ---------------------------------------------------------------------------
// 1. Per-batch coord stats: mean (3) and scale = 2*max||c-mean||  (LDS reduce)
// ---------------------------------------------------------------------------
__global__ __launch_bounds__(256) void stats_kernel(const float* __restrict__ coords,
                                                    float* __restrict__ stats)
{
    const int b = blockIdx.x;
    __shared__ float red[256];
    __shared__ float meansh[3];
    for (int d = 0; d < 3; ++d) {
        float s = 0.f;
        for (int n = threadIdx.x; n < NPTS; n += 256)
            s += coords[((size_t)(b * 3 + d) << 15) + n];
        red[threadIdx.x] = s;
        __syncthreads();
        for (int off = 128; off > 0; off >>= 1) {
            if (threadIdx.x < off) red[threadIdx.x] += red[threadIdx.x + off];
            __syncthreads();
        }
        if (threadIdx.x == 0) meansh[d] = red[0] * (1.0f / (float)NPTS);
        __syncthreads();
    }
    const float m0 = meansh[0], m1 = meansh[1], m2 = meansh[2];
    float mx = 0.f;
    for (int n = threadIdx.x; n < NPTS; n += 256) {
        const float dx = coords[((size_t)(b * 3 + 0) << 15) + n] - m0;
        const float dy = coords[((size_t)(b * 3 + 1) << 15) + n] - m1;
        const float dz = coords[((size_t)(b * 3 + 2) << 15) + n] - m2;
        mx = fmaxf(mx, sqrtf(dx * dx + dy * dy + dz * dz));
    }
    red[threadIdx.x] = mx;
    __syncthreads();
    for (int off = 128; off > 0; off >>= 1) {
        if (threadIdx.x < off) red[threadIdx.x] = fmaxf(red[threadIdx.x], red[threadIdx.x + off]);
        __syncthreads();
    }
    if (threadIdx.x == 0) {
        stats[b * 4 + 0] = m0; stats[b * 4 + 1] = m1; stats[b * 4 + 2] = m2;
        stats[b * 4 + 3] = 2.0f * red[0];
    }
}

// ---------------------------------------------------------------------------
// 2. Scatter-add voxelize: one thread per (point, channel). Coalesced f32
//    atomics. Also emits norm_coords [B,3,N] and voxel counts.
// ---------------------------------------------------------------------------
__global__ __launch_bounds__(256) void scatter_kernel(const float*  __restrict__ coords,
                                                      const __bf16* __restrict__ featT,
                                                      const float*  __restrict__ stats,
                                                      float* __restrict__ sums,
                                                      float* __restrict__ counts,
                                                      float* __restrict__ ncoords)
{
    const int gid = blockIdx.x * 256 + threadIdx.x;      // B*N*64
    const int c   = gid & 63;
    const int pnt = gid >> 6;
    const int b   = pnt >> 15;
    const int n   = pnt & (NPTS - 1);
    const float inv = 1.0f / stats[b * 4 + 3];
    float nrm[3]; int vc[3];
#pragma unroll
    for (int d = 0; d < 3; ++d) {
        const float cd = coords[((size_t)(b * 3 + d) << 15) + n];
        float f = ((cd - stats[b * 4 + d]) * inv + 0.5f) * (float)RR;
        f = fminf(fmaxf(f, 0.0f), (float)(RR - 1));
        nrm[d] = f;
        vc[d]  = (int)rintf(f);                          // round-half-even == jnp.round
    }
    const int idx = ((vc[0] << 5) + vc[1]) * 32 + vc[2];
    atomicAdd(&sums[(((size_t)b << 15) + idx) * 64 + c], (float)featT[((size_t)pnt) * 64 + c]);
    if (c < 3) ncoords[((size_t)(b * 3 + c) << 15) + n] = nrm[c];
    if (c == 0) atomicAdd(&counts[((size_t)b << 15) + idx], 1.0f);
}

// ---------------------------------------------------------------------------
// 3. sums/counts -> bf16 voxel grid [B, R3, 64]
// ---------------------------------------------------------------------------
__global__ __launch_bounds__(256) void voxnorm_kernel(const float* __restrict__ sums,
                                                      const float* __restrict__ counts,
                                                      __bf16* __restrict__ vox)
{
    const size_t gid = (size_t)blockIdx.x * 256 + threadIdx.x;   // B*R3*64
    vox[gid] = (__bf16)(sums[gid] / fmaxf(counts[gid >> 6], 1.0f));
}

// ---------------------------------------------------------------------------
// 4. Weight prepack into exact WMMA B-fragment order:
//    layout [tap][cc][q][lane][j]; K=(lane/16)*8+(j%8)+(j/8)*16, ci=cc*32+K, co=q*16+lane%16
// ---------------------------------------------------------------------------
__global__ __launch_bounds__(256) void prepack_conv_kernel(const float* __restrict__ w,
                                                           __bf16* __restrict__ wpre)
{
    const int gid = blockIdx.x * 256 + threadIdx.x;
    if (gid >= 27 * 2 * 4 * 32 * 16) return;
    const int j    = gid & 15;
    const int lane = (gid >> 4) & 31;
    const int q    = (gid >> 9) & 3;
    const int cc   = (gid >> 11) & 1;
    const int t    = gid >> 12;
    const int k    = ((lane >> 4) << 3) + (j & 7) + ((j >> 3) << 4);
    const int ci   = cc * 32 + k;
    const int co   = (q << 4) + (lane & 15);
    wpre[gid] = (__bf16)w[(size_t)(co * 64 + ci) * 27 + t];
}

__global__ __launch_bounds__(256) void prepack_p_kernel(const float* __restrict__ w,
                                                        __bf16* __restrict__ wpre)
{
    const int gid = blockIdx.x * 256 + threadIdx.x;
    if (gid >= 2 * 4 * 32 * 16) return;
    const int j    = gid & 15;
    const int lane = (gid >> 4) & 31;
    const int q    = (gid >> 9) & 3;
    const int cc   = gid >> 11;
    const int k    = ((lane >> 4) << 3) + (j & 7) + ((j >> 3) << 4);
    const int ci   = cc * 32 + k;
    const int co   = (q << 4) + (lane & 15);
    wpre[gid] = (__bf16)w[co * 64 + ci];
}

// ---------------------------------------------------------------------------
// 5. 3x3x3 conv as 27 shifted WMMA GEMMs.
//    Wave tile: one full z-row = 2 x (16 positions) x 64 out channels
//    (8 accumulators). All 8 waves of a block share x (consecutive y).
//    Per (kx,ky): 24KB weight slab cooperatively staged into LDS; x/y bounds
//    are wave-uniform scalar skips; z edge handled by branchless zero-row
//    pointer select (no exec churn). Inner body: 8 ds_load B + 4 global A
//    loads feeding 8 back-to-back WMMAs.
// ---------------------------------------------------------------------------
__global__ __launch_bounds__(256) void conv3_wmma_kernel(const __bf16* __restrict__ in,
                                                         const __bf16* __restrict__ wpre,
                                                         const float*  __restrict__ bias,
                                                         const __bf16* __restrict__ zrow,
                                                         float* __restrict__ out)
{
    __shared__ __bf16 sw[3 * 2 * 4 * 512];       // 12288 bf16 = 24KB
    const int wave  = threadIdx.x >> 5;
    const int lane  = threadIdx.x & 31;
    const int rowid = blockIdx.x * 8 + wave;     // (b,x,y) z-row; block shares x
    const int b     = rowid >> 10;
    const int x     = (rowid >> 5) & 31;
    const int y     = rowid & 31;
    const int m     = lane & 15;
    const int half  = lane >> 4;
    const size_t bbase = (size_t)b << 15;

    const v8f vzero = {0,0,0,0,0,0,0,0};
    v8f acc[8];                                   // [tile0 q0..3, tile1 q0..3]
#pragma unroll
    for (int i = 0; i < 8; ++i) acc[i] = vzero;

    for (int kx = 0; kx < 3; ++kx) {
        const int xx = x + kx - 1;
        for (int ky = 0; ky < 3; ++ky) {
            const int yy = y + ky - 1;
            // ---- cooperative stage of this (kx,ky) 3-tap weight slab ----
            {
                const u32x4* src = (const u32x4*)(wpre + (size_t)((kx * 3 + ky) * 3) * 4096);
                u32x4* dst = (u32x4*)sw;
                __syncthreads();
#pragma unroll
                for (int k = 0; k < 6; ++k)
                    dst[threadIdx.x + k * 256] = src[threadIdx.x + k * 256];
                __builtin_prefetch(src + 1536, 0, 1);   // next slab
                __syncthreads();
            }
            const bool okxy = ((unsigned)xx < 32u) & ((unsigned)yy < 32u);
            if (!okxy) continue;                  // wave-uniform scalar skip
            const __bf16* bxy = in + (bbase + (size_t)(((xx << 5) + yy) << 5)) * 64;

            for (int kz = 0; kz < 3; ++kz) {
                const int zz0 = m + kz - 1;       // tile0 rows (z = 0..15)
                const int zz1 = 16 + m + kz - 1;  // tile1 rows (z = 16..31)
                const __bf16* r0 = ((unsigned)zz0 < 32u) ? (bxy + zz0 * 64) : zrow;
                const __bf16* r1 = ((unsigned)zz1 < 32u) ? (bxy + zz1 * 64) : zrow;
#pragma unroll
                for (int cc = 0; cc < 2; ++cc) {
                    AF a0, a1;
                    a0.q[0] = *(const u32x4*)(r0 + cc * 32 + half * 8);
                    a0.q[1] = *(const u32x4*)(r0 + cc * 32 + 16 + half * 8);
                    a1.q[0] = *(const u32x4*)(r1 + cc * 32 + half * 8);
                    a1.q[1] = *(const u32x4*)(r1 + cc * 32 + 16 + half * 8);
                    const __bf16* wb = sw + ((kz * 2 + cc) * 4) * 512 + lane * 16;
                    AF b0, b1, b2, b3;
                    b0.q[0] = *(const u32x4*)(wb + 0 * 512); b0.q[1] = *(const u32x4*)(wb + 0 * 512 + 8);
                    b1.q[0] = *(const u32x4*)(wb + 1 * 512); b1.q[1] = *(const u32x4*)(wb + 1 * 512 + 8);
                    b2.q[0] = *(const u32x4*)(wb + 2 * 512); b2.q[1] = *(const u32x4*)(wb + 2 * 512 + 8);
                    b3.q[0] = *(const u32x4*)(wb + 3 * 512); b3.q[1] = *(const u32x4*)(wb + 3 * 512 + 8);
                    acc[0] = __builtin_amdgcn_wmma_f32_16x16x32_bf16(false, a0.v, false, b0.v, (short)0, acc[0], false, false);
                    acc[1] = __builtin_amdgcn_wmma_f32_16x16x32_bf16(false, a0.v, false, b1.v, (short)0, acc[1], false, false);
                    acc[2] = __builtin_amdgcn_wmma_f32_16x16x32_bf16(false, a0.v, false, b2.v, (short)0, acc[2], false, false);
                    acc[3] = __builtin_amdgcn_wmma_f32_16x16x32_bf16(false, a0.v, false, b3.v, (short)0, acc[3], false, false);
                    acc[4] = __builtin_amdgcn_wmma_f32_16x16x32_bf16(false, a1.v, false, b0.v, (short)0, acc[4], false, false);
                    acc[5] = __builtin_amdgcn_wmma_f32_16x16x32_bf16(false, a1.v, false, b1.v, (short)0, acc[5], false, false);
                    acc[6] = __builtin_amdgcn_wmma_f32_16x16x32_bf16(false, a1.v, false, b2.v, (short)0, acc[6], false, false);
                    acc[7] = __builtin_amdgcn_wmma_f32_16x16x32_bf16(false, a1.v, false, b3.v, (short)0, acc[7], false, false);
                }
            }
        }
    }

    // bias + store (D layout: row = v + 8*half, col = q*16 + lane%16)
    const int nlo   = lane & 15;
    const int pbase = ((x << 5) + y) << 5;
#pragma unroll
    for (int mt = 0; mt < 2; ++mt) {
#pragma unroll
        for (int q = 0; q < 4; ++q) {
            const int co = q * 16 + nlo;
            const float bv = bias[co];
#pragma unroll
            for (int v = 0; v < 8; ++v) {
                const int pos = pbase + mt * 16 + v + 8 * half;
                out[(bbase + pos) * 64 + co] = acc[mt * 4 + q][v] + bv;
            }
        }
    }
}

// ---------------------------------------------------------------------------
// 6. Pointwise GEMM branch: pf_pre = wp @ features + bp via WMMA
// ---------------------------------------------------------------------------
__global__ __launch_bounds__(256) void pgemm_wmma_kernel(const __bf16* __restrict__ featT,
                                                         const __bf16* __restrict__ wpre,
                                                         const float*  __restrict__ bias,
                                                         float* __restrict__ out)
{
    const int wave = threadIdx.x >> 5;
    const int lane = threadIdx.x & 31;
    const int tile = blockIdx.x * 8 + wave;
    const int b    = tile >> 11;
    const int n0   = (tile & 2047) << 4;
    const int m    = lane & 15;
    const int half = lane >> 4;
    const size_t bbase = (size_t)b << 15;
    const __bf16* row = featT + (bbase + n0 + m) * 64;

    v8f acc0 = {0,0,0,0,0,0,0,0};
    v8f acc1 = acc0, acc2 = acc0, acc3 = acc0;

#pragma unroll
    for (int cc = 0; cc < 2; ++cc) {
        AF a;
        a.q[0] = *(const u32x4*)(row + cc * 32 + half * 8);
        a.q[1] = *(const u32x4*)(row + cc * 32 + 16 + half * 8);
        const __bf16* wb = wpre + (size_t)(cc * 4) * 512 + (size_t)lane * 16;
        AF b0, b1, b2, b3;
        b0.q[0] = *(const u32x4*)(wb + 0 * 512); b0.q[1] = *(const u32x4*)(wb + 0 * 512 + 8);
        b1.q[0] = *(const u32x4*)(wb + 1 * 512); b1.q[1] = *(const u32x4*)(wb + 1 * 512 + 8);
        b2.q[0] = *(const u32x4*)(wb + 2 * 512); b2.q[1] = *(const u32x4*)(wb + 2 * 512 + 8);
        b3.q[0] = *(const u32x4*)(wb + 3 * 512); b3.q[1] = *(const u32x4*)(wb + 3 * 512 + 8);
        acc0 = __builtin_amdgcn_wmma_f32_16x16x32_bf16(false, a.v, false, b0.v, (short)0, acc0, false, false);
        acc1 = __builtin_amdgcn_wmma_f32_16x16x32_bf16(false, a.v, false, b1.v, (short)0, acc1, false, false);
        acc2 = __builtin_amdgcn_wmma_f32_16x16x32_bf16(false, a.v, false, b2.v, (short)0, acc2, false, false);
        acc3 = __builtin_amdgcn_wmma_f32_16x16x32_bf16(false, a.v, false, b3.v, (short)0, acc3, false, false);
    }

    const int nlo = lane & 15;
    v8f* accs[4] = { &acc0, &acc1, &acc2, &acc3 };
#pragma unroll
    for (int q = 0; q < 4; ++q) {
        const int co = q * 16 + nlo;
        const float bv = bias[co];
#pragma unroll
        for (int v = 0; v < 8; ++v)
            out[(bbase + n0 + v + 8 * half) * 64 + co] = (*accs[q])[v] + bv;
    }
}

// ---------------------------------------------------------------------------
// 7. GroupNorm reduce: sum & sumsq per (b, group) via LDS tree + atomics
//    x is [B, 32768 rows, 64 ch]; grid = B*8*64, 512 rows per block
// ---------------------------------------------------------------------------
__global__ __launch_bounds__(256) void gn_reduce_kernel(const float* __restrict__ x,
                                                        float* __restrict__ acc)
{
    const int blk   = blockIdx.x;
    const int chunk = blk & 63;
    const int g     = (blk >> 6) & 7;
    const int b     = blk >> 9;
    const int ch    = threadIdx.x & 7;
    const int rl    = threadIdx.x >> 3;   // 0..31
    float s = 0.f, ss = 0.f;
    const size_t base = (((size_t)b << 15) + (size_t)chunk * 512) * 64 + g * 8 + ch;
#pragma unroll 4
    for (int i = 0; i < 16; ++i) {
        const float v = x[base + (size_t)(rl + i * 32) * 64];
        s += v; ss += v * v;
    }
    __shared__ float sh1[256], sh2[256];
    sh1[threadIdx.x] = s; sh2[threadIdx.x] = ss;
    __syncthreads();
    for (int off = 128; off > 0; off >>= 1) {
        if (threadIdx.x < off) {
            sh1[threadIdx.x] += sh1[threadIdx.x + off];
            sh2[threadIdx.x] += sh2[threadIdx.x + off];
        }
        __syncthreads();
    }
    if (threadIdx.x == 0) {
        atomicAdd(&acc[(b * 8 + g) * 2 + 0], sh1[0]);
        atomicAdd(&acc[(b * 8 + g) * 2 + 1], sh2[0]);
    }
}

// ---------------------------------------------------------------------------
// 8. GroupNorm apply + swish (bf16 and f32 output variants)
// ---------------------------------------------------------------------------
__device__ __forceinline__ float gn_swish(float xv, const float* __restrict__ acc, int bg,
                                          const float* __restrict__ gamma,
                                          const float* __restrict__ beta, int c)
{
    const float mu  = acc[bg * 2 + 0] * (1.0f / GN_CNT);
    const float var = acc[bg * 2 + 1] * (1.0f / GN_CNT) - mu * mu;
    const float xn  = (xv - mu) * rsqrtf(var + GN_EPS);
    const float v   = xn * gamma[c] + beta[c];
    return v / (1.0f + __expf(-v));
}

__global__ __launch_bounds__(256) void gn_apply_bf16_kernel(const float* __restrict__ x,
                                                            const float* __restrict__ acc,
                                                            const float* __restrict__ gamma,
                                                            const float* __restrict__ beta,
                                                            __bf16* __restrict__ out)
{
    const size_t gid = (size_t)blockIdx.x * 256 + threadIdx.x;
    const int c = (int)(gid & 63);
    const int b = (int)(gid >> 21);                 // row>>15, row = gid>>6
    out[gid] = (__bf16)gn_swish(x[gid], acc, b * 8 + (c >> 3), gamma, beta, c);
}

__global__ __launch_bounds__(256) void gn_apply_f32_kernel(const float* __restrict__ x,
                                                           const float* __restrict__ acc,
                                                           const float* __restrict__ gamma,
                                                           const float* __restrict__ beta,
                                                           float* __restrict__ out)
{
    const size_t gid = (size_t)blockIdx.x * 256 + threadIdx.x;
    const int c = (int)(gid & 63);
    const int b = (int)(gid >> 21);
    out[gid] = gn_swish(x[gid], acc, b * 8 + (c >> 3), gamma, beta, c);
}

// ---------------------------------------------------------------------------
// 9. Trilinear devoxelize + add point branch; coalesced channel-major store
// ---------------------------------------------------------------------------
__global__ __launch_bounds__(256) void devox_final_kernel(const float* __restrict__ h2n,
                                                          const float* __restrict__ pf,
                                                          const float* __restrict__ ncoords,
                                                          float* __restrict__ out)
{
    const int gid = blockIdx.x * 256 + threadIdx.x;  // B*N
    const int b = gid >> 15;
    const int n = gid & (NPTS - 1);
    const float fx = ncoords[((size_t)(b * 3 + 0) << 15) + n];
    const float fy = ncoords[((size_t)(b * 3 + 1) << 15) + n];
    const float fz = ncoords[((size_t)(b * 3 + 2) << 15) + n];
    const int x0 = (int)floorf(fx), y0 = (int)floorf(fy), z0 = (int)floorf(fz);
    const float tx = fx - (float)x0, ty = fy - (float)y0, tz = fz - (float)z0;
    const int xs[2] = { x0, min(x0 + 1, 31) };
    const int ys[2] = { y0, min(y0 + 1, 31) };
    const int zs[2] = { z0, min(z0 + 1, 31) };
    const float wx[2] = { 1.0f - tx, tx };
    const float wy[2] = { 1.0f - ty, ty };
    const float wz[2] = { 1.0f - tz, tz };

    f32x4 acc[16];
    const float* pfp = pf + ((size_t)gid) * 64;
#pragma unroll
    for (int i = 0; i < 16; ++i) acc[i] = *(const f32x4*)(pfp + i * 4);

#pragma unroll
    for (int ix = 0; ix < 2; ++ix)
#pragma unroll
        for (int iy = 0; iy < 2; ++iy)
#pragma unroll
            for (int iz = 0; iz < 2; ++iz) {
                const float w = wx[ix] * wy[iy] * wz[iz];
                const float* src = h2n + (((size_t)b << 15) +
                                          (size_t)(((xs[ix] << 5) + ys[iy]) * 32 + zs[iz])) * 64;
#pragma unroll
                for (int i = 0; i < 16; ++i) {
                    const f32x4 v = *(const f32x4*)(src + i * 4);
                    acc[i] += v * w;
                }
            }

    float* ob = out + (((size_t)b * 64) << 15) + n;
#pragma unroll
    for (int i = 0; i < 16; ++i)
#pragma unroll
        for (int j = 0; j < 4; ++j)
            ob[((size_t)(i * 4 + j)) << 15] = acc[i][j];
}

// ---------------------------------------------------------------------------
// Host side
// ---------------------------------------------------------------------------
extern "C" void kernel_launch(void* const* d_in, const int* in_sizes, int n_in,
                              void* d_out, int out_size, void* d_ws, size_t ws_size,
                              hipStream_t stream)
{
    const float* features = (const float*)d_in[0];
    const float* coords   = (const float*)d_in[1];
    const float* w1  = (const float*)d_in[2];
    const float* b1  = (const float*)d_in[3];
    const float* g1  = (const float*)d_in[4];
    const float* be1 = (const float*)d_in[5];
    const float* w2  = (const float*)d_in[6];
    const float* b2  = (const float*)d_in[7];
    const float* g2  = (const float*)d_in[8];
    const float* be2 = (const float*)d_in[9];
    const float* wp  = (const float*)d_in[10];
    const float* bp  = (const float*)d_in[11];
    const float* gp  = (const float*)d_in[12];
    const float* bep = (const float*)d_in[13];
    float* out = (float*)d_out;

    const size_t BIG = (size_t)BB * R3 * 64;        // 33,554,432 elements
    char* ws = (char*)d_ws;
    size_t off = 0;
    auto alloc = [&](size_t bytes) { char* p = ws + off; off = (off + bytes + 255) & ~(size_t)255; return p; };

    float*  buf0    = (float*) alloc(BIG * 4);              // sums -> h1 -> h2
    float*  counts  = (float*) alloc((size_t)BB * R3 * 4);
    __bf16* vox     = (__bf16*)alloc(BIG * 2);
    __bf16* h1b     = (__bf16*)alloc(BIG * 2);
    float*  h2n     = (float*) alloc(BIG * 4);
    __bf16* featT   = (__bf16*)alloc(BIG * 2);
    float*  pf      = (float*) alloc(BIG * 4);
    float*  ncoords = (float*) alloc((size_t)BB * 3 * NPTS * 4);
    __bf16* wpre1   = (__bf16*)alloc(27 * 2 * 4 * 32 * 16 * 2);
    __bf16* wpre2   = (__bf16*)alloc(27 * 2 * 4 * 32 * 16 * 2);
    __bf16* wppre   = (__bf16*)alloc(2 * 4 * 32 * 16 * 2);
    float*  stats   = (float*) alloc(BB * 4 * 4);
    float*  gnacc   = (float*) alloc(3 * BB * GROUPS * 2 * 4);
    __bf16* zrow    = (__bf16*)alloc(128);                  // 64 zero bf16s
    float* gnacc1 = gnacc;
    float* gnacc2 = gnacc + BB * GROUPS * 2;
    float* gnaccP = gnacc + 2 * BB * GROUPS * 2;

    // zero the atomic accumulation regions + zero row (ws is poisoned)
    hipMemsetAsync(buf0,   0, BIG * 4, stream);
    hipMemsetAsync(counts, 0, (size_t)BB * R3 * 4, stream);
    hipMemsetAsync(gnacc,  0, 3 * BB * GROUPS * 2 * 4, stream);
    hipMemsetAsync(zrow,   0, 128, stream);

    // feature transpose + coord stats + weight prepack
    transpose_kernel<<<dim3(NPTS / 32, 2 * BB), 256, 0, stream>>>(features, featT);
    stats_kernel<<<BB, 256, 0, stream>>>(coords, stats);
    prepack_conv_kernel<<<(27 * 2 * 4 * 32 * 16 + 255) / 256, 256, 0, stream>>>(w1, wpre1);
    prepack_conv_kernel<<<(27 * 2 * 4 * 32 * 16 + 255) / 256, 256, 0, stream>>>(w2, wpre2);
    prepack_p_kernel<<<(2 * 4 * 32 * 16 + 255) / 256, 256, 0, stream>>>(wp, wppre);

    // voxelize
    scatter_kernel<<<(int)(BIG / 256), 256, 0, stream>>>(coords, featT, stats, buf0, counts, ncoords);
    voxnorm_kernel<<<(int)(BIG / 256), 256, 0, stream>>>(buf0, counts, vox);

    // conv1 + GN + swish -> bf16    (grid: B*32*32 z-rows / 8 waves = 2048)
    conv3_wmma_kernel<<<BB * 1024 / 8, 256, 0, stream>>>(vox, wpre1, b1, zrow, buf0);
    gn_reduce_kernel<<<BB * GROUPS * 64, 256, 0, stream>>>(buf0, gnacc1);
    gn_apply_bf16_kernel<<<(int)(BIG / 256), 256, 0, stream>>>(buf0, gnacc1, g1, be1, h1b);

    // conv2 + GN + swish -> f32
    conv3_wmma_kernel<<<BB * 1024 / 8, 256, 0, stream>>>(h1b, wpre2, b2, zrow, buf0);
    gn_reduce_kernel<<<BB * GROUPS * 64, 256, 0, stream>>>(buf0, gnacc2);
    gn_apply_f32_kernel<<<(int)(BIG / 256), 256, 0, stream>>>(buf0, gnacc2, g2, be2, h2n);

    // point branch: GEMM + GN + swish (in place)
    pgemm_wmma_kernel<<<BB * NPTS / 128, 256, 0, stream>>>(featT, wppre, bp, pf);
    gn_reduce_kernel<<<BB * GROUPS * 64, 256, 0, stream>>>(pf, gnaccP);
    gn_apply_f32_kernel<<<(int)(BIG / 256), 256, 0, stream>>>(pf, gnaccP, gp, bep, pf);

    // devoxelize + fuse add, then coords pass-through
    devox_final_kernel<<<BB * NPTS / 256, 256, 0, stream>>>(h2n, pf, ncoords, out);
    hipMemcpyAsync(out + (size_t)BB * 64 * NPTS, coords,
                   (size_t)BB * 3 * NPTS * sizeof(float), hipMemcpyDeviceToDevice, stream);
}
